// LSNNPolicy_70428873720507
// MI455X (gfx1250) — matrix operand order repaired and checked by reading, not compiled, not run on hardware
//
#include <hip/hip_runtime.h>

typedef _Float16 v16h __attribute__((ext_vector_type(16)));
typedef _Float16 v8h  __attribute__((ext_vector_type(8)));
typedef float    v8f  __attribute__((ext_vector_type(8)));

#define TSTEPS 40
#define KW     160            // combined K row stride (halfs): 8 x | 128 z | 24 pad
#define NSLAB  5              // 160 / 32
#define WROWS  144            // 128 hidden cols + 16 padded w_out rows

union H16 { v16h v; _Float16 h[16]; v8h q[2]; };
union F8  { v8f  v; float f[8]; };

// Fused LSNN forward. One block = 16 batch rows, 8 waves; wave w owns hidden
// columns [16w,16w+16). Per step a double-buffered LDS tile holds the combined
// vector [x_t | z_{t-1} | 0]; a 5-slab f16 WMMA chain computes
// x_t@w_in.T + z_{t-1}@w_rec.T with f32 accumulation. Wave 0 reuses the same
// A-fragments against a zero-padded w_out B-fragment for the LI readout.
__global__ __launch_bounds__(256) void
lsnn_fused_kernel(const float* __restrict__ x,
                  const float* __restrict__ w_in,
                  const float* __restrict__ w_rec,
                  const float* __restrict__ w_out,
                  float* __restrict__ out)
{
  __shared__ _Float16 wc[WROWS * KW];    // staged combined weights, f16 (45 KB)
  __shared__ _Float16 zb[2 * 16 * KW];   // double-buffered [x_t | z_{t-1} | 0] (10 KB)
  __shared__ float    mxs[16 * 2];

  const int tid  = threadIdx.x;
  const int lane = tid & 31;
  const int wave = tid >> 5;
  const int hi   = lane >> 4;
  const int nl   = lane & 15;
  const int ncol = wave * 16 + nl;       // hidden column (N of B/C/D)
  const int row0 = blockIdx.x * 16;

  // uniform (scalar) wave predicates -> s_cbranch, EXEC untouched around WMMA
  const bool isw0  = (__builtin_amdgcn_readfirstlane(wave) == 0);
  const bool isenc = (__builtin_amdgcn_readfirstlane(wave) < 4);

  // ---- stage combined weights into LDS (rolled, coalesced) ----
  // row r<128:  [w_in[r][0..7] | w_rec[r][0..127] | 0 x24]
  // row 128+o:  [0 x8          | w_out[o][0..127] | 0 x24]   (o<2, rows 130..143 zero)
  for (int idx = tid; idx < WROWS * KW; idx += 256) {
    const int r = idx / KW, k = idx % KW;
    float w = 0.f;
    if (r < 128) {
      if (k < 8)        w = w_in[r * 8 + k];
      else if (k < 136) w = w_rec[r * 128 + (k - 8)];
    } else if (r < 130) {
      if (k >= 8 && k < 136) w = w_out[(r - 128) * 128 + (k - 8)];
    }
    wc[idx] = (_Float16)w;
  }
  // zero both z-buffers (z_{-1}=0 and permanent pad columns)
  for (int idx = tid; idx < 2 * 16 * KW; idx += 256) zb[idx] = (_Float16)0;

  // ---- encoder constants (threads 0..127: one (row, channel) each) ----
  float ve = 0.f, cc = 0.f;
  const int em = tid >> 3, ech = tid & 7;
  if (isenc) {
    const float sgn = (ech < 4) ? 50.f : -50.f;          // x_pos | x_neg
    cc = sgn * x[(row0 + em) * 4 + (ech & 3)];
    cc = cc > 0.f ? cc : 0.f;                            // relu
  }
  __syncthreads();          // wc + zb zero-fill complete

  if (isenc) {
    ve = 0.1f * cc;                                      // first encoder step
    const float z0 = (ve - 1.f) > 0.f ? 1.f : 0.f;
    zb[em * KW + ech] = (_Float16)z0;                    // x_0 into buf0
    ve *= (1.f - z0);
  }

  // ---- main B fragments from LDS: 2x ds_load_b128 per slab, branch-free ----
  // B order used here (32x16, same as A): lanes 0-15 hold K=0..15 (N=lane&15), lanes 16-31 K=16..31.
  H16 bfrag[NSLAB];
  const _Float16* wrow  = wc + ncol * KW;
  const _Float16* worow = wc + (128 + nl) * KW;          // zeros for nl >= 2
  #pragma unroll
  for (int kk = 0; kk < NSLAB; ++kk) {
    const int ks = kk * 32 + hi * 16;
    bfrag[kk].q[0] = *(const v8h*)(wrow + ks);
    bfrag[kk].q[1] = *(const v8h*)(wrow + ks + 8);
  }

  // ---- state: LSNN C-layout fragment + (wave 0) readout recurrence ----
  float vv[8], ii[8], bb[8], io[8], vo[8], mv[8];
  #pragma unroll
  for (int r = 0; r < 8; ++r) {
    vv[r] = 0.f; ii[r] = 0.f; bb[r] = 1.f;
    io[r] = 0.f; vo[r] = 0.f; mv[r] = -3.0e38f;
  }

  __syncthreads();          // x_0 visible; fragments loaded

  int cur = 0;
  #pragma unroll 1
  for (int t = 0; t < TSTEPS; ++t) {
    const int nxt = cur ^ 1;
    // preload ALL A fragments of [x_t | z_{t-1}]: 10 back-to-back ds_load_b128
    // (16-bit A layout: lane M=nl; h[0..15]=K c0..c0+15, c0 = slab + 16*hi)
    const _Float16* zrow = zb + cur * (16 * KW) + nl * KW;
    H16 a[NSLAB];
    #pragma unroll
    for (int kk = 0; kk < NSLAB; ++kk) {
      const int c0 = kk * 32 + hi * 16;
      a[kk].q[0] = *(const v8h*)(zrow + c0);
      a[kk].q[1] = *(const v8h*)(zrow + c0 + 8);
    }
    // main chain: i-contribution (x and z terms together)
    v8f acc = {0.f, 0.f, 0.f, 0.f, 0.f, 0.f, 0.f, 0.f};
    #pragma unroll
    for (int kk = 0; kk < NSLAB; ++kk)
      acc = __builtin_amdgcn_wmma_f32_16x16x32_f16(false, a[kk].v, false, bfrag[kk].v,
                                                   (short)0, acc, false, false);
    // wave 0 only: LI readout for step s = t-1 (p_{-1}=0 at t=0 is a no-op);
    // single uniform branch, reuses A fragments, w_out B-frags from LDS
    if (isw0) {
      v8f pacc = {0.f, 0.f, 0.f, 0.f, 0.f, 0.f, 0.f, 0.f};
      #pragma unroll
      for (int kk = 0; kk < NSLAB; ++kk) {
        const int ks = kk * 32 + hi * 16;
        H16 wo;
        wo.q[0] = *(const v8h*)(worow + ks);
        wo.q[1] = *(const v8h*)(worow + ks + 8);
        pacc = __builtin_amdgcn_wmma_f32_16x16x32_f16(false, a[kk].v, false, wo.v,
                                                      (short)0, pacc, false, false);
      }
      F8 pf; pf.v = pacc;
      #pragma unroll
      for (int r = 0; r < 8; ++r) {
        vo[r] = vo[r] + 0.1f * (io[r] - vo[r]);   // vo_s uses io_{s-1}
        io[r] = 0.8f * io[r] + pf.f[r];           // io_s
        mv[r] = fmaxf(mv[r], vo[r]);
      }
    }
    // ---- elementwise LSNN cell update ----
    F8 af; af.v = acc;
    _Float16* zwr = zb + nxt * (16 * KW);
    #pragma unroll
    for (int r = 0; r < 8; ++r) {
      const int m  = r + hi * 8;
      const float vd = vv[r] + 0.1f * (ii[r] - vv[r]);
      const float id = 0.8f * ii[r];
      const float bd = bb[r] + 1.4285714e-6f * (1.f - bb[r]);
      const float z  = (vd - bd) > 0.f ? 1.f : 0.f;
      vv[r] = (1.f - z) * vd;
      bb[r] = bd + z * 2.5714286e-3f;
      ii[r] = id + af.f[r];                       // WMMA carries x and z terms
      zwr[m * KW + 8 + ncol] = (_Float16)z;       // publish z_t
    }
    // encoder: produce x_{t+1} into next buffer
    if (isenc) {
      ve += 0.1f * (cc - ve);
      const float zx = (ve - 1.f) > 0.f ? 1.f : 0.f;
      zwr[em * KW + ech] = (_Float16)zx;
      ve *= (1.f - zx);
    }
    __syncthreads();
    cur = nxt;
  }

  // ---- epilogue: readout for final step s = 39 from z_39 in zb[cur] ----
  if (isw0) {
    const _Float16* zrow = zb + cur * (16 * KW) + nl * KW;
    v8f pacc = {0.f, 0.f, 0.f, 0.f, 0.f, 0.f, 0.f, 0.f};
    #pragma unroll
    for (int kk = 0; kk < NSLAB; ++kk) {
      const int c0 = kk * 32 + hi * 16;
      const int ks = kk * 32 + hi * 16;
      H16 a, wo;
      a.q[0]  = *(const v8h*)(zrow + c0);
      a.q[1]  = *(const v8h*)(zrow + c0 + 8);
      wo.q[0] = *(const v8h*)(worow + ks);
      wo.q[1] = *(const v8h*)(worow + ks + 8);
      pacc = __builtin_amdgcn_wmma_f32_16x16x32_f16(false, a.v, false, wo.v,
                                                    (short)0, pacc, false, false);
    }
    F8 pf; pf.v = pacc;
    #pragma unroll
    for (int r = 0; r < 8; ++r) {
      vo[r] = vo[r] + 0.1f * (io[r] - vo[r]);
      io[r] = 0.8f * io[r] + pf.f[r];
      mv[r] = fmaxf(mv[r], vo[r]);
    }
    if (nl < 2) {
      #pragma unroll
      for (int r = 0; r < 8; ++r)
        mxs[(r + 8 * hi) * 2 + nl] = mv[r];       // (row m, class nl)
    }
  }
  __syncthreads();

  // ---- per-row softmax over the 2 classes ----
  if (tid < 16) {
    const float a0 = mxs[tid * 2 + 0];
    const float a1 = mxs[tid * 2 + 1];
    const float mm = fmaxf(a0, a1);
    const float e0 = __expf(a0 - mm);
    const float e1 = __expf(a1 - mm);
    const float s  = e0 + e1;
    out[(row0 + tid) * 2 + 0] = e0 / s;
    out[(row0 + tid) * 2 + 1] = e1 / s;
  }
}

extern "C" void kernel_launch(void* const* d_in, const int* in_sizes, int n_in,
                              void* d_out, int out_size, void* d_ws, size_t ws_size,
                              hipStream_t stream) {
  const float* x     = (const float*)d_in[0];
  const float* w_in  = (const float*)d_in[1];
  const float* w_rec = (const float*)d_in[2];
  const float* w_out = (const float*)d_in[3];
  float* out = (float*)d_out;
  const int B = in_sizes[0] / 4;        // x is (B, 4)
  const int grid = B / 16;              // 16 batch rows per block
  hipLaunchKernelGGL(lsnn_fused_kernel, dim3(grid), dim3(256), 0, stream,
                     x, w_in, w_rec, w_out, out);
}
